// SpeechFeatureExtractor_57681410785540
// MI455X (gfx1250) — matrix operand order repaired and checked by reading, not compiled, not run on hardware
//
#include <hip/hip_runtime.h>
#include <hip/hip_bf16.h>
#include <math.h>

typedef __attribute__((ext_vector_type(16))) _Float16 v16h;
typedef __attribute__((ext_vector_type(8)))  float    v8f;

#define AUDLEN     480000
#define BATCH      16
#define FRAME_LEN  400
#define SHIFT      160
#define NBINS      257     // 512/2 + 1
#define TFRAMES    2998    // 1 + (480000-400)/160
#define NMELS      80
#define K1TILES    13      // K=416 >= 400, tiles of 32
#define FTILES     17      // 272 freq cols >= 257, tiles of 16
#define K2TILES    9       // K=288 >= 257 for mel GEMM
#define MTILES     5       // 80 = 5*16
#define TTILES     188     // ceil(2998/16)
#define TWO_PI_OVER_512 0.012271846303085129f  // 2*pi/512

// ---------------- Setup kernel 1: DFT basis in WMMA B-fragment order -------
// Layout: [part(0=cos,1=-sin)][ftile 0..16][ktile 0..12][lane 0..31][16 halves]
// B-frag (f16 32x16): lanes 0-15 -> N=lane, K=kt*32 + e (e=0..15)
//                     lanes16-31 -> N=lane-16, K=kt*32 + 16 + e
__global__ void build_dft_b(_Float16* __restrict__ dftB) {
    int t = blockIdx.x * blockDim.x + threadIdx.x;
    if (t >= 2 * FTILES * K1TILES * 32) return;
    int lane = t & 31;
    int kt   = (t >> 5) % K1TILES;
    int ft   = ((t >> 5) / K1TILES) % FTILES;
    int p    = (t >> 5) / (K1TILES * FTILES);
    int f    = ft * 16 + (lane & 15);
    int kbase = kt * 32 + ((lane & 16) ? 16 : 0);
    v16h frag;
#pragma unroll
    for (int e = 0; e < 16; ++e) {
        int n = kbase + e;
        float v = 0.0f;
        if (f < NBINS && n < FRAME_LEN) {
            float w = TWO_PI_OVER_512 * (float)f * (float)n;
            v = (p == 0) ? cosf(w) : -sinf(w);
        }
        frag[e] = (_Float16)v;
    }
    *(v16h*)(dftB + (size_t)t * 16) = frag;
}

// ---------------- Setup kernel 2: mel filters in WMMA B-fragment order -----
// Layout: [ntile 0..4][ktile 0..8][lane][16 halves]; N=mel idx, K=freq bin
__global__ void build_mel_b(const float* __restrict__ mel, _Float16* __restrict__ melB) {
    int t = blockIdx.x * blockDim.x + threadIdx.x;
    if (t >= MTILES * K2TILES * 32) return;
    int lane = t & 31;
    int kt   = (t >> 5) % K2TILES;
    int nt   = (t >> 5) / K2TILES;
    int m    = nt * 16 + (lane & 15);
    int kbase = kt * 32 + ((lane & 16) ? 16 : 0);
    v16h frag;
#pragma unroll
    for (int e = 0; e < 16; ++e) {
        int k = kbase + e;
        float v = (k < NBINS) ? mel[m * NBINS + k] : 0.0f;
        frag[e] = (_Float16)v;
    }
    *(v16h*)(melB + (size_t)t * 16) = frag;
}

// ---------------- Main fused kernel: one wave per 16-frame tile ------------
__global__ __launch_bounds__(32)
void melspec_kernel(const float* __restrict__ audio,
                    const float* __restrict__ window,
                    const _Float16* __restrict__ dftB,
                    const _Float16* __restrict__ melB,
                    float* __restrict__ out) {
    __shared__ float    lds_aud[2816];          // staged audio span (incl. x[-1])
    __shared__ float    lds_win[FRAME_LEN];
    __shared__ _Float16 lds_spec[16 * 288];     // |rfft| tile, K-padded to 288

    const int tile = blockIdx.x;                // 0..187 frame tile
    const int b    = blockIdx.y;                // batch
    const int lane = threadIdx.x;               // 0..31

    const float* aud = audio + (size_t)b * AUDLEN;
    const long span = (long)tile * (16 * SHIFT) - 1;   // -1 for pre-emphasis

    // Stage audio span (2801 floats) with boundary clamp-to-zero.
    for (int i = lane; i < 2801; i += 32) {
        long g = span + i;
        lds_aud[i] = (g >= 0 && g < AUDLEN) ? aud[g] : 0.0f;
    }
    for (int i = lane; i < FRAME_LEN; i += 32) lds_win[i] = window[i];
    // Zero the stage-2 K-pad columns 272..287 of the spec matrix.
    for (int i = lane; i < 16 * 16; i += 32)
        lds_spec[(i >> 4) * 288 + 272 + (i & 15)] = (_Float16)0.0f;
    __syncthreads();

    // ---- Build A fragments: windowed, pre-emphasized frames (f16 16x32) ----
    // A-frag layout: row = lane&15; kbase = (lane<16)?0:16;
    //   e in 0..15 -> K = kbase + e
    const int row   = lane & 15;
    const int kbA   = (lane < 16) ? 0 : 16;
    v16h afrag[K1TILES];
#pragma unroll
    for (int kt = 0; kt < K1TILES; ++kt) {
#pragma unroll
        for (int e = 0; e < 16; ++e) {
            int k = kbA + e;
            int n = kt * 32 + k;
            float v = 0.0f;
            if (n < FRAME_LEN) {
                int s = row * SHIFT + n;       // local sample; lds_aud shifted by +1
                v = (lds_aud[s + 1] - 0.97f * lds_aud[s]) * lds_win[n];
            }
            afrag[kt][e] = (_Float16)v;
        }
    }

    // C/D layout: VGPR v -> row (v + crow), col = lane&15 within n-tile.
    const int crow = (lane < 16) ? 0 : 8;
    const int ccol = lane & 15;

    // ---- Stage 1: DFT magnitude via paired cos/sin WMMA GEMMs ----
    for (int ft = 0; ft < FTILES; ++ft) {
        v8f accR = {};
        v8f accI = {};
#pragma unroll
        for (int kt = 0; kt < K1TILES; ++kt) {
            v16h br = *(const v16h*)(dftB + ((((size_t)0 * FTILES + ft) * K1TILES + kt) * 32 + lane) * 16);
            v16h bi = *(const v16h*)(dftB + ((((size_t)1 * FTILES + ft) * K1TILES + kt) * 32 + lane) * 16);
            accR = __builtin_amdgcn_wmma_f32_16x16x32_f16(false, afrag[kt], false, br,
                                                          (short)0, accR, false, false);
            accI = __builtin_amdgcn_wmma_f32_16x16x32_f16(false, afrag[kt], false, bi,
                                                          (short)0, accI, false, false);
        }
        int col = ft * 16 + ccol;              // < 272; cols >=257 are zero (zero B)
#pragma unroll
        for (int v = 0; v < 8; ++v) {
            float s = sqrtf(accR[v] * accR[v] + accI[v] * accI[v]);
            lds_spec[(crow + v) * 288 + col] = (_Float16)s;
        }
    }
    __syncthreads();

    // ---- Build stage-2 A fragments from the spec matrix in LDS ----
    v16h a2[K2TILES];
#pragma unroll
    for (int kt = 0; kt < K2TILES; ++kt) {
#pragma unroll
        for (int e = 0; e < 16; ++e) {
            int k = kbA + e;
            a2[kt][e] = lds_spec[row * 288 + kt * 32 + k];
        }
    }

    // ---- Stage 2: mel projection GEMM + log, store ----
    for (int nt = 0; nt < MTILES; ++nt) {
        v8f acc = {};
#pragma unroll
        for (int kt = 0; kt < K2TILES; ++kt) {
            v16h bm = *(const v16h*)(melB + (((size_t)nt * K2TILES + kt) * 32 + lane) * 16);
            acc = __builtin_amdgcn_wmma_f32_16x16x32_f16(false, a2[kt], false, bm,
                                                         (short)0, acc, false, false);
        }
        int m = nt * 16 + ccol;                // < 80 always (80 = 5*16)
#pragma unroll
        for (int v = 0; v < 8; ++v) {
            int fr = tile * 16 + crow + v;
            if (fr < TFRAMES)
                out[((size_t)b * TFRAMES + fr) * NMELS + m] = logf(acc[v] + 1e-6f);
        }
    }
}

extern "C" void kernel_launch(void* const* d_in, const int* in_sizes, int n_in,
                              void* d_out, int out_size, void* d_ws, size_t ws_size,
                              hipStream_t stream) {
    const float* audio  = (const float*)d_in[0];   // [16, 480000] f32
    const float* window = (const float*)d_in[1];   // [400] f32
    const float* mel    = (const float*)d_in[2];   // [80, 257] f32
    float* out = (float*)d_out;                    // [16, 2998, 80] f32

    // Workspace: DFT B-fragments (452,608 B) then mel B-fragments (46,080 B).
    _Float16* dftB = (_Float16*)d_ws;
    _Float16* melB = (_Float16*)((char*)d_ws + (size_t)2 * FTILES * K1TILES * 32 * 16 * sizeof(_Float16));

    {
        int n = 2 * FTILES * K1TILES * 32;     // 14144 threads
        build_dft_b<<<(n + 127) / 128, 128, 0, stream>>>(dftB);
    }
    {
        int n = MTILES * K2TILES * 32;         // 1440 threads
        build_mel_b<<<(n + 127) / 128, 128, 0, stream>>>(mel, melB);
    }
    dim3 grid(TTILES, BATCH);                  // 188 x 16 tiles, one wave each
    melspec_kernel<<<grid, 32, 0, stream>>>(audio, window, dftB, melB, out);
}
